// GCN_11106785427525
// MI455X (gfx1250) — matrix operand order repaired and checked by reading, not compiled, not run on hardware
//
#include <hip/hip_runtime.h>
#include <hip/hip_bf16.h>

#define N_NODES 50000
#define N_EDGES 800000
#define X_DIM   128
#define H_DIM   256
#define Y_DIM   128

typedef __attribute__((ext_vector_type(16))) __bf16 v16bf;
typedef __attribute__((ext_vector_type(8)))  float  v8f;

// ---------------------------------------------------------------------------
// CDNA5 async global->LDS copy (ASYNCcnt-tracked), per CDNA5 ISA: each lane
// copies 16B; VDST VGPR holds the LDS byte address (= low 32 bits of generic
// pointer), VADDR the 64-bit global address.
// ---------------------------------------------------------------------------
__device__ __forceinline__ void async_load_b128(unsigned lds_off, const void* gaddr) {
    asm volatile("global_load_async_to_lds_b128 %0, %1, off"
                 :: "v"(lds_off), "v"(gaddr) : "memory");
}
__device__ __forceinline__ void wait_asynccnt0() {
    asm volatile("s_wait_asynccnt 0x0" ::: "memory");
}

// ---------------------------------------------------------------------------
// Kernel 1: row-normalize x (fp32 [N,128]) -> xn (bf16 [N,128]).
// One wave per row; each lane owns 4 contiguous floats.
// ---------------------------------------------------------------------------
__global__ void row_norm_bf16(const float* __restrict__ x,
                              __bf16* __restrict__ xn, int nRows) {
    int wid  = (blockIdx.x * blockDim.x + threadIdx.x) >> 5;
    int lane = threadIdx.x & 31;
    if (wid >= nRows) return;
    const float4 xv = ((const float4*)(x + (size_t)wid * X_DIM))[lane];
    float s = xv.x + xv.y + xv.z + xv.w;
    #pragma unroll
    for (int off = 16; off > 0; off >>= 1) s += __shfl_xor(s, off, 32);
    float inv = 1.0f / (s + 1e-4f);
    union { __bf16 h[4]; uint2 u; } o;
    o.h[0] = (__bf16)(xv.x * inv);
    o.h[1] = (__bf16)(xv.y * inv);
    o.h[2] = (__bf16)(xv.z * inv);
    o.h[3] = (__bf16)(xv.w * inv);
    *(uint2*)(xn + (size_t)wid * X_DIM + lane * 4) = o.u;
}

// ---------------------------------------------------------------------------
// One-shot weight transpose + convert: WT[n*K + k] = (bf16) W[k*N + n].
// ---------------------------------------------------------------------------
__global__ void transpose_w_bf16(const float* __restrict__ W,
                                 __bf16* __restrict__ WT, int K, int N) {
    int idx = blockIdx.x * blockDim.x + threadIdx.x;
    if (idx >= K * N) return;
    int n = idx / K;
    int k = idx - n * K;
    WT[idx] = (__bf16)W[(size_t)k * N + n];
}

// ---------------------------------------------------------------------------
// SpMM scatter: one wave per edge; coalesced bf16 row gather, scale, native
// fp32 global atomics into dense accumulator. Working set is L2-resident.
// ---------------------------------------------------------------------------
__global__ void spmm_scatter_d128(const float* __restrict__ vals,
                                  const int* __restrict__ rows,
                                  const int* __restrict__ cols,
                                  const __bf16* __restrict__ X,
                                  float* __restrict__ out, int nE) {
    int wid  = (blockIdx.x * blockDim.x + threadIdx.x) >> 5;
    int lane = threadIdx.x & 31;
    if (wid >= nE) return;
    float v = vals[wid];
    int r = rows[wid], c = cols[wid];
    union { uint2 u; __bf16 h[4]; } ld;
    ld.u = *(const uint2*)(X + (size_t)c * 128 + lane * 4);
    float* op = out + (size_t)r * 128 + lane * 4;
    #pragma unroll
    for (int j = 0; j < 4; ++j)
        __hip_atomic_fetch_add(op + j, v * (float)ld.h[j],
                               __ATOMIC_RELAXED, __HIP_MEMORY_SCOPE_AGENT);
}

__global__ void spmm_scatter_d256(const float* __restrict__ vals,
                                  const int* __restrict__ rows,
                                  const int* __restrict__ cols,
                                  const __bf16* __restrict__ X,
                                  float* __restrict__ out, int nE) {
    int wid  = (blockIdx.x * blockDim.x + threadIdx.x) >> 5;
    int lane = threadIdx.x & 31;
    if (wid >= nE) return;
    float v = vals[wid];
    int r = rows[wid], c = cols[wid];
    union { uint4 u; __bf16 h[8]; } ld;
    ld.u = *(const uint4*)(X + (size_t)c * 256 + lane * 8);
    float* op = out + (size_t)r * 256 + lane * 8;
    #pragma unroll
    for (int j = 0; j < 8; ++j)
        __hip_atomic_fetch_add(op + j, v * (float)ld.h[j],
                               __ATOMIC_RELAXED, __HIP_MEMORY_SCOPE_AGENT);
}

// ---------------------------------------------------------------------------
// WMMA A fragment (wave32), contiguous k order matching the B fragment:
// lane L holds row M = L%16; elem i -> K = kb + (L>=16?16:0) + i.
// fp32 source, cvt to bf16.
// ---------------------------------------------------------------------------
__device__ __forceinline__ v16bf load_a_frag(const float* __restrict__ A,
                                             int lda, int row0, int kb, int lane) {
    int m    = lane & 15;
    int koff = (lane >> 4) << 4;  // 0 or 16
    const float* rp = A + (size_t)(row0 + m) * lda + kb;
    v16bf a;
    #pragma unroll
    for (int i = 0; i < 8; ++i) a[i]     = (__bf16)rp[koff + i];
    #pragma unroll
    for (int i = 0; i < 8; ++i) a[8 + i] = (__bf16)rp[koff + 8 + i];
    return a;
}

// ---------------------------------------------------------------------------
// GEMM: C[M,N] = A[M,K] @ W[K,N] + bias, W pre-transposed bf16 (WT[N,K]).
// Block = 8 waves = 8 row tiles x one 64-col group.
//  * B strip (64 x KDIM bf16, contiguous in WT) staged into LDS via
//    global_load_async_to_lds_b128 + s_wait_asynccnt + barrier.
//  * Each wave: A fragment once per K-step, 4 back-to-back wmmas against
//    LDS-resident B fragments (ds_load_b128 pairs, 32B-aligned).
// B frag: lane L holds col N=L%16, elem i -> K = kb + (L>=16?16:0) + i.
// C/D: acc[r] is (M = (L>=16?8:0)+r, N = L%16).
// ---------------------------------------------------------------------------
template <int KDIM, bool RELU, bool OUT_BF16>
__global__ void wmma_gemm_bias(const float* __restrict__ A,
                               const __bf16* __restrict__ WT,
                               const float* __restrict__ bias,
                               void* __restrict__ Cv,
                               int M, int N) {
    constexpr int CT   = 4;
    constexpr int COLS = 16 * CT;                    // 64
    __shared__ __bf16 smemB[COLS * KDIM];

    int cgroups = N / COLS;
    int rb = blockIdx.x / cgroups;                   // row block (8 tiles)
    int tg = blockIdx.x - rb * cgroups;              // column group

    // --- stage B strip: 64*KDIM*2 bytes, contiguous in WT ------------------
    {
        const int BYTES = COLS * KDIM * 2;           // 16KB or 32KB
        unsigned    lbase = (unsigned)(size_t)&smemB[0];
        const char* gbase = (const char*)(WT + (size_t)tg * COLS * KDIM);
        for (int off = threadIdx.x * 16; off < BYTES; off += 256 * 16)
            async_load_b128(lbase + off, gbase + off);
        wait_asynccnt0();
        __syncthreads();
    }

    int lane  = threadIdx.x & 31;
    int tm    = rb * 8 + (threadIdx.x >> 5);
    if (tm >= (M >> 4)) return;
    int row0  = tm << 4;

    int nlo   = lane & 15;
    int koffB = (lane >> 4) << 4;                    // 0 or 16

    v8f acc[CT];
    #pragma unroll
    for (int c = 0; c < CT; ++c) acc[c] = (v8f){};

    for (int kb = 0; kb < KDIM; kb += 32) {
        v16bf a = load_a_frag(A, KDIM, row0, kb, lane);
        v16bf b[CT];
        #pragma unroll
        for (int c = 0; c < CT; ++c)
            b[c] = *(const v16bf*)(smemB + (c * 16 + nlo) * KDIM + kb + koffB);
        #pragma unroll
        for (int c = 0; c < CT; ++c)
            acc[c] = __builtin_amdgcn_wmma_f32_16x16x32_bf16(
                false, a, false, b[c], (short)0, acc[c], false, false);
    }

    int mbase = (lane >> 4) << 3;
    #pragma unroll
    for (int c = 0; c < CT; ++c) {
        int col = tg * COLS + c * 16 + nlo;
        float bv = bias[col];
        #pragma unroll
        for (int r = 0; r < 8; ++r) {
            float v = acc[c][r] + bv;
            if (RELU) v = v > 0.0f ? v : 0.0f;
            size_t idx = (size_t)(row0 + mbase + r) * N + col;
            if (OUT_BF16) ((__bf16*)Cv)[idx] = (__bf16)v;
            else          ((float*)Cv)[idx]  = v;
        }
    }
}

// ---------------------------------------------------------------------------
// Launch. Inputs: x, adj_vals, adj_row, adj_col, W1, b1, W2, b2.
// Workspace layout (256B-aligned, total ~115.33 MB):
//   xn   bf16 [N,128]   @ 0         | agg1 f32 [N,128] @ 12.8 MB
//   h    bf16 [N,256]   @ 38.4 MB   | agg2 f32 [N,256] @ 64.0 MB
//   WT1  bf16 [256,128] @ 115.2 MB  | WT2 bf16 [128,256] @ +64 KB
// ---------------------------------------------------------------------------
extern "C" void kernel_launch(void* const* d_in, const int* in_sizes, int n_in,
                              void* d_out, int out_size, void* d_ws, size_t ws_size,
                              hipStream_t stream) {
    const float* x        = (const float*)d_in[0];
    const float* adj_vals = (const float*)d_in[1];
    const int*   adj_row  = (const int*)d_in[2];
    const int*   adj_col  = (const int*)d_in[3];
    const float* W1       = (const float*)d_in[4];
    const float* b1       = (const float*)d_in[5];
    const float* W2       = (const float*)d_in[6];
    const float* b2       = (const float*)d_in[7];

    char*   ws   = (char*)d_ws;
    __bf16* xn   = (__bf16*)(ws);
    float*  agg1 = (float*)(ws + 12800000);
    __bf16* h    = (__bf16*)(ws + 38400000);
    float*  agg2 = (float*)(ws + 64000000);
    __bf16* WT1  = (__bf16*)(ws + 115200000);
    __bf16* WT2  = (__bf16*)(ws + 115265536);

    const int BLK = 256;                 // 8 wave32 waves per block
    const int WPB = BLK / 32;

    // 1) normalize rows -> bf16; transpose+convert weights
    row_norm_bf16<<<(N_NODES + WPB - 1) / WPB, BLK, 0, stream>>>(x, xn, N_NODES);
    transpose_w_bf16<<<(X_DIM * H_DIM + BLK - 1) / BLK, BLK, 0, stream>>>(
        W1, WT1, X_DIM, H_DIM);
    transpose_w_bf16<<<(H_DIM * Y_DIM + BLK - 1) / BLK, BLK, 0, stream>>>(
        W2, WT2, H_DIM, Y_DIM);

    // 2) zero accumulators (memset nodes are graph-capture legal)
    hipMemsetAsync(agg1, 0, (size_t)N_NODES * X_DIM * sizeof(float), stream);
    hipMemsetAsync(agg2, 0, (size_t)N_NODES * H_DIM * sizeof(float), stream);

    // 3) spmm1: agg1 = A @ xn
    spmm_scatter_d128<<<(N_EDGES + WPB - 1) / WPB, BLK, 0, stream>>>(
        adj_vals, adj_row, adj_col, xn, agg1, N_EDGES);

    // 4) gemm1: h = relu(agg1 @ W1 + b1) -> bf16
    {
        int rbs  = (N_NODES / 16 + 7) / 8;           // 391 row blocks
        int grid = rbs * (H_DIM / 64);               // * 4 col groups
        wmma_gemm_bias<X_DIM, true, true><<<grid, BLK, 0, stream>>>(
            agg1, WT1, b1, (void*)h, N_NODES, H_DIM);
    }

    // 5) spmm2: agg2 = A @ h
    spmm_scatter_d256<<<(N_EDGES + WPB - 1) / WPB, BLK, 0, stream>>>(
        adj_vals, adj_row, adj_col, h, agg2, N_EDGES);

    // 6) gemm2: y = agg2 @ W2 + b2 -> f32 out
    {
        int rbs  = (N_NODES / 16 + 7) / 8;           // 391 row blocks
        int grid = rbs * (Y_DIM / 64);               // * 2 col groups
        wmma_gemm_bias<H_DIM, false, false><<<grid, BLK, 0, stream>>>(
            agg2, WT2, b2, d_out, N_NODES, Y_DIM);
    }
}
